// SSLSchNetModel_34093450396361
// MI455X (gfx1250) — matrix operand order repaired and checked by reading, not compiled, not run on hardware
//
#include <hip/hip_runtime.h>

typedef __attribute__((ext_vector_type(16))) _Float16 v16h;
typedef __attribute__((ext_vector_type(8)))  _Float16 v8h;
typedef __attribute__((ext_vector_type(8)))  float    v8f;

#define DIMX 64
#define NC 50
static __device__ __constant__ float GAPC = 5.0f / 49.0f;   // center spacing
static __device__ __constant__ float COEF = -49.0f / 5.0f;  // -1/gap

__device__ __forceinline__ float softplus_b05(float x) {
  // torch Softplus(beta=0.5, threshold=14): linear when 0.5x > 14
  return (x > 28.0f) ? x : 2.0f * logf(1.0f + __expf(0.5f * x));
}

__device__ __forceinline__ v8f wmma16(v16h a, v16h b, v8f c) {
  return __builtin_amdgcn_wmma_f32_16x16x32_f16(false, a, false, b, (short)0, c,
                                                false, false);
}

__device__ __forceinline__ v8f vzero8() {
  v8f z;
#pragma unroll
  for (int p = 0; p < 8; ++p) z[p] = 0.0f;
  return z;
}

// B fragment: weights stored row-major [n][64] f16; per-lane two 8-runs of K (k = 8h+s, 16+8h+s).
__device__ __forceinline__ v16h load_bfrag(const _Float16* __restrict__ Bw,
                                           int l15, int half, int nt, int kk) {
  const _Float16* p = Bw + (nt * 16 + l15) * 64 + kk * 32 + half * 8;
  v8h lo = *(const v8h*)p;
  v8h hi = *(const v8h*)(p + 16);
  v16h b;
#pragma unroll
  for (int e = 0; e < 8; ++e) {
    b[e] = lo[e];
    b[e + 8] = hi[e];
  }
  return b;
}

// A fragment from 16 f32 (two contiguous 8-runs) at X[rowoff + ...].
__device__ __forceinline__ v16h afrag_from_f32(const float* __restrict__ X,
                                               unsigned rowoff, int half,
                                               int kk) {
  const unsigned kb = rowoff + (unsigned)(kk * 32 + half * 8);
  v16h a;
#pragma unroll
  for (int o = 0; o < 8; ++o) {
    a[o]     = (_Float16)X[kb + o];
    a[o + 8] = (_Float16)X[kb + 16 + o];
  }
  return a;
}

// ---------------------------------------------------------------------------
// Weight prep: f32 -> f16, cf_w1 padded 50 -> 64 along K.
// ---------------------------------------------------------------------------
__global__ void prep_weights_kernel(const float* __restrict__ conv_w1,
                                    const float* __restrict__ cf_w1,
                                    const float* __restrict__ cf_w2,
                                    const float* __restrict__ nl2_w,
                                    const float* __restrict__ nl3_w,
                                    _Float16* __restrict__ Cw,
                                    _Float16* __restrict__ B1w,
                                    _Float16* __restrict__ B2w,
                                    _Float16* __restrict__ NL2w,
                                    _Float16* __restrict__ NL3w) {
  int g = blockIdx.x * blockDim.x + threadIdx.x;
  if (g >= 3 * 64 * 64) return;
  int k = g & 63, n = (g >> 6) & 63, i = g >> 12;
  Cw[g]   = (_Float16)conv_w1[g];
  B2w[g]  = (_Float16)cf_w2[g];
  NL2w[g] = (_Float16)nl2_w[g];
  NL3w[g] = (_Float16)nl3_w[g];
  B1w[g]  = (_Float16)((k < NC) ? cf_w1[(i * 64 + n) * NC + k] : 0.0f);
}

__global__ void mask_set_kernel(const int* __restrict__ sel,
                                unsigned char* __restrict__ masked, int K) {
  int g = blockIdx.x * blockDim.x + threadIdx.x;
  if (g < K) masked[sel[g]] = 1;
}

__global__ void node_init_kernel(const float* __restrict__ emb,
                                 const int* __restrict__ ntype,
                                 float* __restrict__ node, int total) {
  int g = blockIdx.x * blockDim.x + threadIdx.x;
  if (g >= total) return;
  int nid = g >> 6, k = g & 63;
  node[g] = emb[(unsigned)ntype[nid] * 64u + (unsigned)k];
}

// ---------------------------------------------------------------------------
// new_node = node @ conv_w1.T  ([rows,64] x [64,64]) via WMMA, 16 rows/wave.
// ---------------------------------------------------------------------------
__global__ __launch_bounds__(256) void gemm_node_kernel(
    const float* __restrict__ X, const _Float16* __restrict__ Bw,
    float* __restrict__ Y, int rows) {
  const int lane = threadIdx.x & 31;
  const int wav  = threadIdx.x >> 5;
  const int half = (lane >> 4) & 1;
  const int l15  = lane & 15;

  v16h bf[2][4];
#pragma unroll
  for (int kk = 0; kk < 2; ++kk)
#pragma unroll
    for (int nt = 0; nt < 4; ++nt) bf[kk][nt] = load_bfrag(Bw, l15, half, nt, kk);

  const int ntiles = rows >> 4;
  const int nw = gridDim.x * 8;
  for (int tile = blockIdx.x * 8 + wav; tile < ntiles; tile += nw) {
    const unsigned rowoff = (unsigned)(tile * 16 + l15) * 64u;
    v8f acc[4];
#pragma unroll
    for (int nt = 0; nt < 4; ++nt) acc[nt] = vzero8();
#pragma unroll
    for (int kk = 0; kk < 2; ++kk) {
      v16h a = afrag_from_f32(X, rowoff, half, kk);
#pragma unroll
      for (int nt = 0; nt < 4; ++nt) acc[nt] = wmma16(a, bf[kk][nt], acc[nt]);
    }
#pragma unroll
    for (int nt = 0; nt < 4; ++nt)
#pragma unroll
      for (int p = 0; p < 8; ++p) {
        unsigned o =
            (unsigned)(tile * 16 + p + half * 8) * 64u + (unsigned)(nt * 16 + l15);
        Y[o] = acc[nt][p];
      }
  }
}

// ---------------------------------------------------------------------------
// Fused edge pipeline per 16-edge tile (one wave):
//   rbf (recomputed) -> WMMA(cf_w1) -> softplus -> LDS reshape ->
//   WMMA(cf_w2) -> msg = new_node[src]*h -> atomicAdd agg[dst]
// ---------------------------------------------------------------------------
__global__ __launch_bounds__(256) void edge_conv_kernel(
    const float* __restrict__ dist, const unsigned char* __restrict__ masked,
    const float* __restrict__ edge_mask, const _Float16* __restrict__ B1w,
    const float* __restrict__ b1, const _Float16* __restrict__ B2w,
    const float* __restrict__ b2, const float* __restrict__ new_node,
    const int* __restrict__ src, const int* __restrict__ dst,
    float* __restrict__ agg, int E) {
  __shared__ __attribute__((aligned(32))) _Float16 smem[8 * 16 * 64];
  const int lane = threadIdx.x & 31;
  const int wav  = threadIdx.x >> 5;
  const int half = (lane >> 4) & 1;
  const int l15  = lane & 15;
  _Float16* lds = smem + wav * (16 * 64);

  v16h b1f[2][4], b2f[2][4];
#pragma unroll
  for (int kk = 0; kk < 2; ++kk)
#pragma unroll
    for (int nt = 0; nt < 4; ++nt) {
      b1f[kk][nt] = load_bfrag(B1w, l15, half, nt, kk);
      b2f[kk][nt] = load_bfrag(B2w, l15, half, nt, kk);
    }
  float b1v[4], b2v[4];
#pragma unroll
  for (int nt = 0; nt < 4; ++nt) {
    b1v[nt] = b1[nt * 16 + l15];
    b2v[nt] = b2[nt * 16 + l15];
  }

  const int ntiles = E >> 4;
  const int nw = gridDim.x * 8;
  for (int tile = blockIdx.x * 8 + wav; tile < ntiles; tile += nw) {
    const int e0 = tile << 4;
    const int myedge = e0 + l15;  // this lane's A-matrix row (both half-waves)
    const float d = dist[myedge];
    const bool flg = masked[myedge] != 0;

    // ---- stage 1: t1 = rbf @ cf_w1.T ----
    v8f acc[4];
#pragma unroll
    for (int nt = 0; nt < 4; ++nt) acc[nt] = vzero8();
#pragma unroll
    for (int kk = 0; kk < 2; ++kk) {
      v16h a;
#pragma unroll
      for (int e = 0; e < 16; ++e) {
        int k = kk * 32 + ((e >> 3) << 4) + half * 8 + (e & 7);
        float val = 0.0f;
        if (k < NC) {
          if (flg) {
            val = edge_mask[k];
          } else {
            float t = d - (float)k * GAPC;
            val = __expf(COEF * t * t);
          }
        }
        a[e] = (_Float16)val;
      }
#pragma unroll
      for (int nt = 0; nt < 4; ++nt) acc[nt] = wmma16(a, b1f[kk][nt], acc[nt]);
    }

    // ---- softplus + reshape D-layout -> A-layout through LDS ----
#pragma unroll
    for (int nt = 0; nt < 4; ++nt)
#pragma unroll
      for (int p = 0; p < 8; ++p) {
        int m = p + half * 8;
        lds[m * 64 + nt * 16 + l15] =
            (_Float16)softplus_b05(acc[nt][p] + b1v[nt]);
      }

    // ---- stage 2: h = sp(t1) @ cf_w2.T ----
    v8f acc2[4];
#pragma unroll
    for (int nt = 0; nt < 4; ++nt) acc2[nt] = vzero8();
#pragma unroll
    for (int kk = 0; kk < 2; ++kk) {
      const int kb = kk * 32 + half * 8;
      v8h lo = *(const v8h*)(lds + l15 * 64 + kb);
      v8h hi = *(const v8h*)(lds + l15 * 64 + kb + 16);
      v16h a;
#pragma unroll
      for (int e = 0; e < 8; ++e) {
        a[e] = lo[e];
        a[e + 8] = hi[e];
      }
#pragma unroll
      for (int nt = 0; nt < 4; ++nt) acc2[nt] = wmma16(a, b2f[kk][nt], acc2[nt]);
    }

    // ---- msg = new_node[src] * h ; scatter-add into agg[dst] (L2 atomics) --
    const int base8 = e0 + half * 8;
    int s8[8], t8[8];
    *(int4*)&s8[0] = *(const int4*)(src + base8);
    *(int4*)&s8[4] = *(const int4*)(src + base8 + 4);
    *(int4*)&t8[0] = *(const int4*)(dst + base8);
    *(int4*)&t8[4] = *(const int4*)(dst + base8 + 4);
#pragma unroll
    for (int p = 0; p < 8; ++p) {
      const unsigned so = (unsigned)s8[p] * 64u;
      const unsigned to = (unsigned)t8[p] * 64u;
#pragma unroll
      for (int nt = 0; nt < 4; ++nt) {
        const unsigned n = (unsigned)(nt * 16 + l15);
        float h = acc2[nt][p] + b2v[nt];
        unsafeAtomicAdd((float*)&agg[to + n], new_node[so + n] * h);
      }
    }
  }
}

// ---------------------------------------------------------------------------
// node += sp(agg @ nl2.T + b2) @ nl3.T + b3   (fused 2-stage WMMA, 16 rows/wave)
// ---------------------------------------------------------------------------
__global__ __launch_bounds__(256) void node_update_kernel(
    const float* __restrict__ agg, const _Float16* __restrict__ NL2w,
    const float* __restrict__ nl2_b, const _Float16* __restrict__ NL3w,
    const float* __restrict__ nl3_b, float* __restrict__ node, int rows) {
  __shared__ __attribute__((aligned(32))) _Float16 smem[8 * 16 * 64];
  const int lane = threadIdx.x & 31;
  const int wav  = threadIdx.x >> 5;
  const int half = (lane >> 4) & 1;
  const int l15  = lane & 15;
  _Float16* lds = smem + wav * (16 * 64);

  v16h bf2[2][4], bf3[2][4];
#pragma unroll
  for (int kk = 0; kk < 2; ++kk)
#pragma unroll
    for (int nt = 0; nt < 4; ++nt) {
      bf2[kk][nt] = load_bfrag(NL2w, l15, half, nt, kk);
      bf3[kk][nt] = load_bfrag(NL3w, l15, half, nt, kk);
    }
  float b2v[4], b3v[4];
#pragma unroll
  for (int nt = 0; nt < 4; ++nt) {
    b2v[nt] = nl2_b[nt * 16 + l15];
    b3v[nt] = nl3_b[nt * 16 + l15];
  }

  const int ntiles = rows >> 4;
  const int nw = gridDim.x * 8;
  for (int tile = blockIdx.x * 8 + wav; tile < ntiles; tile += nw) {
    const unsigned rowoff = (unsigned)(tile * 16 + l15) * 64u;
    v8f acc[4];
#pragma unroll
    for (int nt = 0; nt < 4; ++nt) acc[nt] = vzero8();
#pragma unroll
    for (int kk = 0; kk < 2; ++kk) {
      v16h a = afrag_from_f32(agg, rowoff, half, kk);
#pragma unroll
      for (int nt = 0; nt < 4; ++nt) acc[nt] = wmma16(a, bf2[kk][nt], acc[nt]);
    }
#pragma unroll
    for (int nt = 0; nt < 4; ++nt)
#pragma unroll
      for (int p = 0; p < 8; ++p) {
        int m = p + half * 8;
        lds[m * 64 + nt * 16 + l15] =
            (_Float16)softplus_b05(acc[nt][p] + b2v[nt]);
      }
    v8f acc2[4];
#pragma unroll
    for (int nt = 0; nt < 4; ++nt) acc2[nt] = vzero8();
#pragma unroll
    for (int kk = 0; kk < 2; ++kk) {
      const int kb = kk * 32 + half * 8;
      v8h lo = *(const v8h*)(lds + l15 * 64 + kb);
      v8h hi = *(const v8h*)(lds + l15 * 64 + kb + 16);
      v16h a;
#pragma unroll
      for (int e = 0; e < 8; ++e) {
        a[e] = lo[e];
        a[e + 8] = hi[e];
      }
#pragma unroll
      for (int nt = 0; nt < 4; ++nt) acc2[nt] = wmma16(a, bf3[kk][nt], acc2[nt]);
    }
#pragma unroll
    for (int nt = 0; nt < 4; ++nt)
#pragma unroll
      for (int p = 0; p < 8; ++p) {
        unsigned idx =
            (unsigned)(tile * 16 + p + half * 8) * 64u + (unsigned)(nt * 16 + l15);
        node[idx] += acc2[nt][p] + b3v[nt];
      }
  }
}

// ---------------------------------------------------------------------------
// Output heads (tiny GEMMs; VALU with LDS-cached weights)
// ---------------------------------------------------------------------------
__global__ __launch_bounds__(256) void node_head_kernel(
    const float* __restrict__ node, const int* __restrict__ idx,
    const float* __restrict__ w1, const float* __restrict__ b1,
    const float* __restrict__ w2, const float* __restrict__ b2,
    float* __restrict__ out, int nsel) {
  __shared__ float sw1[32 * 64];
  __shared__ float sb1[32];
  __shared__ float sw2[3 * 32];
  __shared__ float sb2[3];
  for (int i = threadIdx.x; i < 32 * 64; i += blockDim.x) sw1[i] = w1[i];
  for (int i = threadIdx.x; i < 32; i += blockDim.x) sb1[i] = b1[i];
  for (int i = threadIdx.x; i < 96; i += blockDim.x) sw2[i] = w2[i];
  for (int i = threadIdx.x; i < 3; i += blockDim.x) sb2[i] = b2[i];
  __syncthreads();
  int t = blockIdx.x * blockDim.x + threadIdx.x;
  if (t >= nsel) return;
  const unsigned roff = (unsigned)idx[t] * 64u;
  float rr[64];
#pragma unroll
  for (int k = 0; k < 64; ++k) rr[k] = node[roff + k];
  float o0 = sb2[0], o1 = sb2[1], o2 = sb2[2];
  for (int j = 0; j < 32; ++j) {
    float h = sb1[j];
#pragma unroll
    for (int k = 0; k < 64; ++k) h += rr[k] * sw1[j * 64 + k];
    o0 += sw2[j] * h;
    o1 += sw2[32 + j] * h;
    o2 += sw2[64 + j] * h;
  }
  out[t * 3 + 0] = o0;
  out[t * 3 + 1] = o1;
  out[t * 3 + 2] = o2;
}

__global__ __launch_bounds__(256) void edge_head_kernel(
    const float* __restrict__ node, const int* __restrict__ sidx,
    const int* __restrict__ tidx, const float* __restrict__ w1,
    const float* __restrict__ b1, const float* __restrict__ w2,
    const float* __restrict__ b2, float* __restrict__ out, int esel) {
  __shared__ float sw1[64 * 128];
  __shared__ float sb1[64];
  __shared__ float sw2[5 * 64];
  __shared__ float sb2[5];
  for (int i = threadIdx.x; i < 64 * 128; i += blockDim.x) sw1[i] = w1[i];
  for (int i = threadIdx.x; i < 64; i += blockDim.x) sb1[i] = b1[i];
  for (int i = threadIdx.x; i < 5 * 64; i += blockDim.x) sw2[i] = w2[i];
  for (int i = threadIdx.x; i < 5; i += blockDim.x) sb2[i] = b2[i];
  __syncthreads();
  int t = blockIdx.x * blockDim.x + threadIdx.x;
  if (t >= esel) return;
  const unsigned soff = (unsigned)sidx[t] * 64u;
  const unsigned toff = (unsigned)tidx[t] * 64u;
  float rr[128];
#pragma unroll
  for (int k = 0; k < 64; ++k) {
    rr[k] = node[soff + k];
    rr[64 + k] = node[toff + k];
  }
  float o[5];
#pragma unroll
  for (int q = 0; q < 5; ++q) o[q] = sb2[q];
  for (int j = 0; j < 64; ++j) {
    float h = sb1[j];
#pragma unroll
    for (int k = 0; k < 128; ++k) h += rr[k] * sw1[j * 128 + k];
#pragma unroll
    for (int q = 0; q < 5; ++q) o[q] += sw2[q * 64 + j] * h;
  }
#pragma unroll
  for (int q = 0; q < 5; ++q) out[t * 5 + q] = o[q];
}

// ---------------------------------------------------------------------------
extern "C" void kernel_launch(void* const* d_in, const int* in_sizes, int n_in,
                              void* d_out, int out_size, void* d_ws,
                              size_t ws_size, hipStream_t stream) {
  (void)n_in;
  (void)out_size;
  (void)ws_size;
  const float* emb       = (const float*)d_in[0];
  const float* dist      = (const float*)d_in[1];
  const float* edge_mask = (const float*)d_in[2];
  const float* conv_w1   = (const float*)d_in[3];
  const float* cf_w1     = (const float*)d_in[4];
  const float* cf_b1     = (const float*)d_in[5];
  const float* cf_w2     = (const float*)d_in[6];
  const float* cf_b2     = (const float*)d_in[7];
  const float* nl2_w     = (const float*)d_in[8];
  const float* nl2_b     = (const float*)d_in[9];
  const float* nl3_w     = (const float*)d_in[10];
  const float* nl3_b     = (const float*)d_in[11];
  const float* nt1_w     = (const float*)d_in[12];
  const float* nt1_b     = (const float*)d_in[13];
  const float* nt2_w     = (const float*)d_in[14];
  const float* nt2_b     = (const float*)d_in[15];
  const float* et1_w     = (const float*)d_in[16];
  const float* et1_b     = (const float*)d_in[17];
  const float* et2_w     = (const float*)d_in[18];
  const float* et2_b     = (const float*)d_in[19];
  const int* node_type   = (const int*)d_in[20];
  const int* src         = (const int*)d_in[21];
  const int* dst         = (const int*)d_in[22];
  const int* sel         = (const int*)d_in[23];
  const int* node_index  = (const int*)d_in[24];
  const int* source_idx  = (const int*)d_in[25];
  const int* target_idx  = (const int*)d_in[26];

  const int N    = in_sizes[20];
  const int E    = in_sizes[1];
  const int K    = in_sizes[23];
  const int NSEL = in_sizes[24];
  const int ESEL = in_sizes[25];

  // workspace carving (256B aligned)
  char* ws = (char*)d_ws;
  size_t off = 0;
  auto carve = [&](size_t bytes) -> char* {
    char* p = ws + off;
    off = (off + bytes + 255) & ~(size_t)255;
    return p;
  };
  float* node     = (float*)carve((size_t)N * 64 * sizeof(float));
  float* new_node = (float*)carve((size_t)N * 64 * sizeof(float));
  float* agg      = (float*)carve((size_t)N * 64 * sizeof(float));
  unsigned char* masked = (unsigned char*)carve((size_t)E);
  _Float16* Cw   = (_Float16*)carve(3 * 4096 * sizeof(_Float16));
  _Float16* B1w  = (_Float16*)carve(3 * 4096 * sizeof(_Float16));
  _Float16* B2w  = (_Float16*)carve(3 * 4096 * sizeof(_Float16));
  _Float16* NL2w = (_Float16*)carve(3 * 4096 * sizeof(_Float16));
  _Float16* NL3w = (_Float16*)carve(3 * 4096 * sizeof(_Float16));

  float* out_node = (float*)d_out;
  float* out_edge = out_node + (size_t)NSEL * 3;

  hipMemsetAsync(masked, 0, (size_t)E, stream);
  prep_weights_kernel<<<(3 * 64 * 64 + 255) / 256, 256, 0, stream>>>(
      conv_w1, cf_w1, cf_w2, nl2_w, nl3_w, Cw, B1w, B2w, NL2w, NL3w);
  mask_set_kernel<<<(K + 255) / 256, 256, 0, stream>>>(sel, masked, K);
  node_init_kernel<<<(N * 64 + 255) / 256, 256, 0, stream>>>(emb, node_type,
                                                             node, N * 64);

  for (int i = 0; i < 3; ++i) {
    gemm_node_kernel<<<256, 256, 0, stream>>>(node, Cw + i * 4096, new_node, N);
    hipMemsetAsync(agg, 0, (size_t)N * 64 * sizeof(float), stream);
    edge_conv_kernel<<<2048, 256, 0, stream>>>(
        dist, masked, edge_mask, B1w + i * 4096, cf_b1 + i * 64,
        B2w + i * 4096, cf_b2 + i * 64, new_node, src, dst, agg, E);
    node_update_kernel<<<256, 256, 0, stream>>>(agg, NL2w + i * 4096,
                                                nl2_b + i * 64, NL3w + i * 4096,
                                                nl3_b + i * 64, node, N);
  }

  node_head_kernel<<<(NSEL + 255) / 256, 256, 0, stream>>>(
      node, node_index, nt1_w, nt1_b, nt2_w, nt2_b, out_node, NSEL);
  edge_head_kernel<<<(ESEL + 255) / 256, 256, 0, stream>>>(
      node, source_idx, target_idx, et1_w, et1_b, et2_w, et2_b, out_edge, ESEL);
}
